// KnowledgeCircuit_57423712747575
// MI455X (gfx1250) — matrix operand order repaired and checked by reading, not compiled, not run on hardware
//
#include <hip/hip_runtime.h>
#include <cstdint>
#include <cstddef>

// ---------------- problem constants (from reference) ----------------
#define BB    4
#define SS    4096
#define TTOT  (BB*SS)      // 16384 tokens
#define DDIM  512
#define NNEU  1024
#define MEXP  128
#define RR    64
// tile config
#define TM    64           // tokens per workgroup
#define WAVES 4            // 4 wave32 = 128 threads
#define XROW  (DDIM + 8)   // padded bf16 row (1040B -> bank stride 4)
#define GROW  (MEXP + 1)   // padded f32 gate row (129 dwords -> bank stride 1)
#define HROW  (RR + 4)     // padded f32 h row (68 dwords -> bank stride 4)

typedef __attribute__((ext_vector_type(16))) __bf16       bf16x16;
typedef __attribute__((ext_vector_type(8)))  float        f32x8;
typedef __attribute__((ext_vector_type(4)))  unsigned int u32x4;

union ABFrag {
    bf16x16 v;
    u32x4   q[2];
    __bf16  e[16];
};

__device__ __forceinline__ f32x8 zero8() {
    f32x8 z;
#pragma unroll
    for (int i = 0; i < 8; ++i) z[i] = 0.0f;
    return z;
}

// --------------------------------------------------------------------
// P1: gate [T,1024] -> gate_b [T,128] : sum groups of 8, normalize over M
// --------------------------------------------------------------------
__global__ void gate_group_kernel(const float* __restrict__ gate,
                                  float* __restrict__ gate_b) {
    const int t = blockIdx.x;
    const int m = threadIdx.x;           // 0..127
    const float* g = gate + (size_t)t * NNEU + m * 8;
    float s = 0.0f;
#pragma unroll
    for (int j = 0; j < 8; ++j) s += g[j];
    __shared__ float red[MEXP];
    red[m] = s;
    __syncthreads();
    for (int off = MEXP / 2; off > 0; off >>= 1) {
        if (m < off) red[m] += red[m + off];
        __syncthreads();
    }
    const float tot = red[0];
    gate_b[(size_t)t * MEXP + m] = s / (tot + 1e-8f);
}

// --------------------------------------------------------------------
// P2: x f32 -> bf16 (row-major, same layout)
// --------------------------------------------------------------------
__global__ void cvt_x_kernel(const float* __restrict__ x,
                             __bf16* __restrict__ xb, int n) {
    int i = blockIdx.x * blockDim.x + threadIdx.x;
    const int stride = gridDim.x * blockDim.x;
    for (; i < n; i += stride) xb[i] = (__bf16)x[i];
}

__device__ __forceinline__ unsigned int pack_bf16x2(float f0, float f1) {
    __bf16 b0 = (__bf16)f0;
    __bf16 b1 = (__bf16)f1;
    unsigned short s0 = __builtin_bit_cast(unsigned short, b0);
    unsigned short s1 = __builtin_bit_cast(unsigned short, b1);
    return (unsigned int)s0 | ((unsigned int)s1 << 16);
}

// --------------------------------------------------------------------
// P3: know_f [M,512,64] -> WMMA-B fragment tiles.
// Tile (n, kb in 0..15, rb in 0..3): 32 lanes x 32B.
// Lane L covers K = kb*32 + (L<16?0:16) + 2j(,+1), column r = rb*16 + L%16.
// --------------------------------------------------------------------
__global__ void pack_kf_kernel(const float* __restrict__ kf,
                               unsigned int* __restrict__ out) {
    const int id = blockIdx.x * blockDim.x + threadIdx.x;   // 2,097,152 total
    if (id >= MEXP * 16 * 4 * 32 * 8) return;
    const int j    = id & 7;
    const int lane = (id >> 3) & 31;
    const int rb   = (id >> 8) & 3;
    const int kb   = (id >> 10) & 15;
    const int n    = id >> 14;
    const int k = kb * 32 + ((lane < 16) ? 0 : 16) + 2 * j;
    const int r = rb * 16 + (lane & 15);
    const float* p = kf + ((size_t)n * DDIM + k) * RR + r;
    const int tile = (n * 16 + kb) * 4 + rb;
    out[(size_t)tile * 256 + lane * 8 + j] = pack_bf16x2(p[0], p[RR]);
}

// --------------------------------------------------------------------
// P4: know_r [M,64,512] -> WMMA-B fragment tiles.
// Tile (n, kb in 0..1, db in 0..31): K = kb*32 + ..., column d = db*16 + L%16.
// --------------------------------------------------------------------
__global__ void pack_kr_kernel(const float* __restrict__ kr,
                               unsigned int* __restrict__ out) {
    const int id = blockIdx.x * blockDim.x + threadIdx.x;   // 2,097,152 total
    if (id >= MEXP * 2 * 32 * 32 * 8) return;
    const int j    = id & 7;
    const int lane = (id >> 3) & 31;
    const int db   = (id >> 8) & 31;
    const int kb   = (id >> 13) & 1;
    const int n    = id >> 14;
    const int k = kb * 32 + ((lane < 16) ? 0 : 16) + 2 * j;
    const int d = db * 16 + (lane & 15);
    const float* p = kr + ((size_t)n * RR + k) * DDIM + d;
    const int tile = (n * 2 + kb) * 32 + db;
    out[(size_t)tile * 256 + lane * 8 + j] = pack_bf16x2(p[0], p[DDIM]);
}

// --------------------------------------------------------------------
// Fused main kernel: per 64-token tile
//   pass A: h = sum_n g_n * (X @ Kf_n)           (bf16 WMMA, f32 accum)
//   pass B: out = sum_n (g_n * h) @ Kr_n         (bf16 WMMA, f32 accum)
// --------------------------------------------------------------------
__launch_bounds__(WAVES * 32)
__global__ void fused_circuit_kernel(const __bf16* __restrict__ xb,
                                     const float* __restrict__ gate_b,
                                     const unsigned int* __restrict__ kf,
                                     const unsigned int* __restrict__ kr,
                                     float* __restrict__ out) {
    extern __shared__ char smem[];
    __bf16* lds_x = (__bf16*)smem;                                   // TM*XROW*2
    float*  lds_g = (float*)(smem + TM * XROW * 2);                  // TM*GROW*4
    float*  lds_h = (float*)(smem + TM * XROW * 2 + TM * GROW * 4);  // TM*HROW*4

    const int tid  = threadIdx.x;
    const int wave = tid >> 5;
    const int lane = tid & 31;
    const int l16  = lane & 15;
    const int lh   = lane >> 4;       // lane half: 0 or 1
    const int row0 = wave * 16;       // this wave's 16 token rows in tile
    const int t0   = blockIdx.x * TM;

    // ---- stage X tile (64 x 512 bf16) into padded LDS ----
    {
        const u32x4* src = (const u32x4*)(xb + (size_t)t0 * DDIM);   // 8 bf16/chunk
        for (int c = tid; c < TM * (DDIM / 8); c += WAVES * 32) {
            const int row = c >> 6;          // 64 chunks per row
            const int col = c & 63;
            u32x4 v = src[c];
            *(u32x4*)(lds_x + row * XROW + col * 8) = v;
        }
    }
    // ---- stage gate tile (64 x 128 f32), padded ----
    {
        const float* gsrc = gate_b + (size_t)t0 * MEXP;
        for (int c = tid; c < TM * MEXP; c += WAVES * 32) {
            const int row = c >> 7;
            const int col = c & 127;
            lds_g[row * GROW + col] = gsrc[c];
        }
    }
    __syncthreads();

    // =================== pass A ===================
    f32x8 h[4];
#pragma unroll
    for (int rb = 0; rb < 4; ++rb) h[rb] = zero8();

    const char* xrow_base = (const char*)(lds_x + (row0 + l16) * XROW);

    for (int n = 0; n < MEXP; n += 2) {
        f32x8 a0[4], a1[4];
#pragma unroll
        for (int rb = 0; rb < 4; ++rb) { a0[rb] = zero8(); a1[rb] = zero8(); }

        if (n + 2 < MEXP) {  // warm L0/L2 for next expert pair -> global_prefetch_b8
            __builtin_prefetch(kf + (size_t)((n + 2) * 16) * 4 * 256, 0, 1);
            __builtin_prefetch(kf + (size_t)((n + 3) * 16) * 4 * 256, 0, 1);
        }

#pragma unroll 4
        for (int kb = 0; kb < 16; ++kb) {
            // A fragment: 16x32 bf16, contiguous K layout (k = 16*lh + s), two ds b128 reads
            ABFrag a;
            const char* ap = xrow_base + (size_t)(kb * 32 + lh * 16) * 2;
            a.q[0] = *(const u32x4*)ap;
            a.q[1] = *(const u32x4*)(ap + 16);
#pragma unroll
            for (int rb = 0; rb < 4; ++rb) {
                ABFrag b0, b1;
                const u32x4* p0 = (const u32x4*)(kf + (size_t)((n * 16 + kb) * 4 + rb) * 256 + lane * 8);
                const u32x4* p1 = (const u32x4*)(kf + (size_t)(((n + 1) * 16 + kb) * 4 + rb) * 256 + lane * 8);
                b0.q[0] = p0[0]; b0.q[1] = p0[1];
                b1.q[0] = p1[0]; b1.q[1] = p1[1];
                a0[rb] = __builtin_amdgcn_wmma_f32_16x16x32_bf16(
                    false, a.v, false, b0.v, (short)0, a0[rb], false, false);
                a1[rb] = __builtin_amdgcn_wmma_f32_16x16x32_bf16(
                    false, a.v, false, b1.v, (short)0, a1[rb], false, false);
            }
        }
        // fold H_n into h with per-token gate (C layout: VGPR j -> row j / j+8)
#pragma unroll
        for (int j = 0; j < 8; ++j) {
            const int tr = row0 + j + lh * 8;
            const float g0 = lds_g[tr * GROW + n];
            const float g1 = lds_g[tr * GROW + n + 1];
#pragma unroll
            for (int rb = 0; rb < 4; ++rb)
                h[rb][j] += g0 * a0[rb][j] + g1 * a1[rb][j];
        }
    }

    // h (C layout) -> LDS f32 [64][68]
#pragma unroll
    for (int rb = 0; rb < 4; ++rb)
#pragma unroll
        for (int j = 0; j < 8; ++j) {
            const int tr = row0 + j + lh * 8;
            lds_h[tr * HROW + rb * 16 + l16] = h[rb][j];
        }
    __syncthreads();

    // =================== pass B ===================
    const float* hrow = lds_h + (row0 + l16) * HROW;
    const float* grow = lds_g + (row0 + l16) * GROW;

    for (int dc = 0; dc < 4; ++dc) {          // 4 chunks of 128 output columns
        f32x8 acc[8];
#pragma unroll
        for (int dt = 0; dt < 8; ++dt) acc[dt] = zero8();

        for (int n = 0; n < MEXP; ++n) {
            const float g = grow[n];
            if (n + 1 < MEXP)
                __builtin_prefetch(kr + (size_t)((n + 1) * 2) * 32 * 256 + (size_t)dc * 8 * 256, 0, 1);
#pragma unroll
            for (int kb = 0; kb < 2; ++kb) {
                // build A fragment = g * h row, f32 -> bf16, contiguous K layout (k = 16*lh + s)
                ABFrag a;
                const float* hp = hrow + kb * 32 + lh * 16;
#pragma unroll
                for (int u = 0; u < 8; ++u) {
                    a.e[u]     = (__bf16)(g * hp[u]);
                    a.e[8 + u] = (__bf16)(g * hp[8 + u]);
                }
#pragma unroll
                for (int dt = 0; dt < 8; ++dt) {
                    ABFrag b;
                    const u32x4* p = (const u32x4*)(kr +
                        (size_t)((n * 2 + kb) * 32 + dc * 8 + dt) * 256 + lane * 8);
                    b.q[0] = p[0]; b.q[1] = p[1];
                    acc[dt] = __builtin_amdgcn_wmma_f32_16x16x32_bf16(
                        false, a.v, false, b.v, (short)0, acc[dt], false, false);
                }
            }
        }
        // store: rows coalesce across lanes 0..15 (64B contiguous)
#pragma unroll
        for (int dt = 0; dt < 8; ++dt)
#pragma unroll
            for (int j = 0; j < 8; ++j) {
                const int tr = row0 + j + lh * 8;
                out[(size_t)(t0 + tr) * DDIM + dc * 128 + dt * 16 + l16] = acc[dt][j];
            }
    }
}

// --------------------------------------------------------------------
extern "C" void kernel_launch(void* const* d_in, const int* in_sizes, int n_in,
                              void* d_out, int out_size, void* d_ws, size_t ws_size,
                              hipStream_t stream) {
    const float* x    = (const float*)d_in[0];   // [B,S,D]   f32
    const float* gate = (const float*)d_in[1];   // [B,S,N]   f32
    const float* knf  = (const float*)d_in[2];   // [M,D,R]   f32
    const float* knr  = (const float*)d_in[3];   // [M,R,D]   f32
    float* out = (float*)d_out;                  // [B,S,D]   f32

    // workspace carve-up (40 MB total, 8MB-aligned offsets)
    char* ws = (char*)d_ws;
    float*        gate_b = (float*)ws;                                   // 8 MB
    __bf16*       xb     = (__bf16*)(ws + (size_t)(8u << 20));           // 16 MB
    unsigned int* kf     = (unsigned int*)(ws + (size_t)(24u << 20));    // 8 MB
    unsigned int* kr     = (unsigned int*)(ws + (size_t)(32u << 20));    // 8 MB

    gate_group_kernel<<<TTOT, MEXP, 0, stream>>>(gate, gate_b);
    cvt_x_kernel<<<2048, 256, 0, stream>>>(x, xb, TTOT * DDIM);
    pack_kf_kernel<<<(MEXP * 16 * 4 * 32 * 8) / 256, 256, 0, stream>>>(knf, kf);
    pack_kr_kernel<<<(MEXP * 2 * 32 * 32 * 8) / 256, 256, 0, stream>>>(knr, kr);

    const size_t lds_bytes = (size_t)TM * XROW * 2   // x tile   (66,560)
                           + (size_t)TM * GROW * 4   // gate     (33,024)
                           + (size_t)TM * HROW * 4;  // h        (17,408)
    fused_circuit_kernel<<<TTOT / TM, WAVES * 32, lds_bytes, stream>>>(
        xb, gate_b, kf, kr, out);
}
